// Block_44452911514382
// MI455X (gfx1250) — compile-verified, matrix operand order checked by reading, not run on hardware
//
#include <hip/hip_runtime.h>
#include <math.h>

// ---------------------------------------------------------------------------
// CDNA5 (gfx1250) wave32 WMMA transformer block.
// GEMMs: C[M,N] = A[M,K](bf16,row-major) x Bt[N,K](bf16, B transposed),
// fp32 accumulate via v_wmma_f32_16x16x32_bf16, tiles staged in LDS with
// double-buffered global_load_async_to_lds_b128 (ASYNCcnt).
// ---------------------------------------------------------------------------

typedef __attribute__((ext_vector_type(16))) __bf16 v16bf;
typedef __attribute__((ext_vector_type(8)))  __bf16 v8bf;
typedef __attribute__((ext_vector_type(8)))  float  v8f;

// Load one 16x32 bf16 fragment (A layout; also B layout when source is Bt).
// ISA layout: lanes 0-15 row M=lane hold K=[0..7],[16..23];
//             lanes 16-31 row M=lane-16 hold K=[8..15],[24..31].
__device__ __forceinline__ v16bf ld_frag(const __bf16* __restrict__ base,
                                         int row_stride) {
  const int lane = threadIdx.x & 31;
  const int half = lane >> 4;
  const int r    = lane & 15;
  const __bf16* p = base + (size_t)r * row_stride + half * 8;
  v8bf lo = *(const v8bf*)(p);
  v8bf hi = *(const v8bf*)(p + 16);
  return __builtin_shufflevector(lo, hi, 0, 1, 2, 3, 4, 5, 6, 7,
                                         8, 9, 10, 11, 12, 13, 14, 15);
}

__device__ __forceinline__ v8f wmma_bf16(v16bf a, v16bf b, v8f c) {
  // 8 args: (neg_a, A, neg_b, B, c_mod, C, reuse_a, reuse_b)
  return __builtin_amdgcn_wmma_f32_16x16x32_bf16(false, a, false, b,
                                                 (short)0, c, false, false);
}

// Padded LDS tile geometry: [128 rows][64 k] bf16, row stride 72 elem (144 B,
// 16B aligned, bank-spread).
#define LDSTR 72

// Cooperative async copy of a 128x64 bf16 tile into LDS (1024 16B chunks,
// 4 per thread with 256 threads). Tracked by ASYNCcnt.
__device__ __forceinline__ void async_copy_tile(const __bf16* g_base,
                                                int src_stride,
                                                __bf16* lds_base) {
  const int tid = threadIdx.x;
  const uint32_t l0 = (uint32_t)(size_t)lds_base;
#pragma unroll
  for (int i = 0; i < 4; ++i) {
    const int c    = i * 256 + tid;   // chunk index, lanes contiguous
    const int row  = c >> 3;          // 8 chunks (128 B) per row
    const int part = c & 7;
    const uint64_t g = (uint64_t)(size_t)(g_base + (size_t)row * src_stride +
                                          part * 8);
    const uint32_t l = l0 + (uint32_t)(row * (LDSTR * 2) + part * 16);
    asm volatile("global_load_async_to_lds_b128 %0, %1, off"
                 :: "v"(l), "v"(g)
                 : "memory");
  }
}

// ---------------------------------------------------------------------------
// Weight prep: dst[N][K](bf16) = src[K][N](f32)   (transpose + convert)
// ---------------------------------------------------------------------------
__global__ void k_convT(const float* __restrict__ src, __bf16* __restrict__ dst,
                        int K, int N) {
  size_t i = (size_t)blockIdx.x * 256 + threadIdx.x;
  if (i >= (size_t)K * N) return;
  size_t n = i / K, k = i % K;
  dst[n * K + k] = (__bf16)src[k * N + n];
}

__global__ void k_cat_bias(const float* __restrict__ bq,
                           const float* __restrict__ bk,
                           const float* __restrict__ bv,
                           float* __restrict__ dst) {
  int i = threadIdx.x;
  if (i < 128)      dst[i] = bq[i];
  else if (i < 256) dst[i] = bk[i - 128];
  else if (i < 384) dst[i] = bv[i - 256];
}

// ---------------------------------------------------------------------------
// LayerNorm: one block per row of D elements. Writes optional fp32 + bf16.
// ---------------------------------------------------------------------------
__global__ __launch_bounds__(256) void k_layernorm(
    const float* __restrict__ x, const float* __restrict__ g,
    const float* __restrict__ b, float* __restrict__ outF,
    __bf16* __restrict__ outB, int D) {
  __shared__ float s1[256];
  __shared__ float s2[256];
  const int row = blockIdx.x;
  const float* xr = x + (size_t)row * D;
  float sum = 0.f, sq = 0.f;
  for (int i = threadIdx.x; i < D; i += 256) {
    float v = xr[i];
    sum += v;
    sq  += v * v;
  }
  s1[threadIdx.x] = sum;
  s2[threadIdx.x] = sq;
  __syncthreads();
  for (int off = 128; off > 0; off >>= 1) {
    if ((int)threadIdx.x < off) {
      s1[threadIdx.x] += s1[threadIdx.x + off];
      s2[threadIdx.x] += s2[threadIdx.x + off];
    }
    __syncthreads();
  }
  const float mu  = s1[0] / (float)D;
  const float var = s2[0] / (float)D - mu * mu;
  const float inv = rsqrtf(var + 1e-5f);
  for (int i = threadIdx.x; i < D; i += 256) {
    float v = (xr[i] - mu) * inv * g[i] + b[i];
    if (outF) outF[(size_t)row * D + i] = v;
    outB[(size_t)row * D + i] = (__bf16)v;
  }
}

// ---------------------------------------------------------------------------
// WMMA GEMM. Block = 256 threads = 8 waves, C tile = 128x128.
// K staged 64-deep through double-buffered LDS tiles filled with
// global_load_async_to_lds_b128; s_wait_asynccnt + barrier per K-tile.
// Wave w computes rows [blk.x*128 + 16w, +16), cols [blk.y*128, +128).
// MODE 0: +qkv_bias; scatter q,k row-major bf16, v transposed bf16
// MODE 1: +bias, +fp32 residual -> fp32 out (may alias residual)
// MODE 2: +bias, exact GELU -> bf16 out
// MODE 3: +bias, +fp32 residual -> fp32 out (final)
// ---------------------------------------------------------------------------
template <int MODE>
__global__ __launch_bounds__(256) void k_gemm(
    const __bf16* __restrict__ A, const __bf16* __restrict__ Bt,
    int M, int N, int K,
    const float* __restrict__ bias, const float* __restrict__ resid,
    float* __restrict__ outF, __bf16* __restrict__ out0,
    __bf16* __restrict__ out1, __bf16* __restrict__ out2, int Sseq) {
  __shared__ __bf16 sA[2][128 * LDSTR];
  __shared__ __bf16 sB[2][128 * LDSTR];

  const int wave   = threadIdx.x >> 5;
  const int lane   = threadIdx.x & 31;
  const int half   = lane >> 4;
  const int lanelo = lane & 15;
  const int r0 = blockIdx.x * 128;
  const int c0 = blockIdx.y * 128;
  const int nkt = K >> 6;  // K tiles of 64

  // prologue: fill buffer 0
  async_copy_tile(A + (size_t)r0 * K, K, &sA[0][0]);
  async_copy_tile(Bt + (size_t)c0 * K, K, &sB[0][0]);

  v8f acc[8] = {};
  for (int kt = 0; kt < nkt; ++kt) {
    // our async writes for buf(kt) done, then block-wide visibility
    asm volatile("s_wait_asynccnt 0x0" ::: "memory");
    __syncthreads();
    // after the barrier nobody reads buf(kt^1) anymore -> refill it
    if (kt + 1 < nkt) {
      async_copy_tile(A + (size_t)r0 * K + (kt + 1) * 64, K,
                      &sA[(kt + 1) & 1][0]);
      async_copy_tile(Bt + (size_t)c0 * K + (kt + 1) * 64, K,
                      &sB[(kt + 1) & 1][0]);
    }
    const __bf16* a_ld = &sA[kt & 1][wave * 16 * LDSTR];
    const __bf16* b_ld = &sB[kt & 1][0];
#pragma unroll
    for (int kk = 0; kk < 2; ++kk) {
      v16bf a = ld_frag(a_ld + kk * 32, LDSTR);
#pragma unroll
      for (int t = 0; t < 8; ++t) {
        v16bf b = ld_frag(b_ld + (size_t)(16 * t) * LDSTR + kk * 32, LDSTR);
        acc[t] = wmma_bf16(a, b, acc[t]);
      }
    }
  }

  const int r0w = r0 + wave * 16;
#pragma unroll
  for (int t = 0; t < 8; ++t) {
#pragma unroll
    for (int j = 0; j < 8; ++j) {
      const int row = r0w + j + 8 * half;
      const int col = c0 + 16 * t + lanelo;
      float v = acc[t][j] + bias[col];
      const size_t idx = (size_t)row * N + col;
      if (MODE == 0) {
        if (col < 128) {
          out0[(size_t)row * 128 + col] = (__bf16)v;              // q
        } else if (col < 256) {
          out1[(size_t)row * 128 + (col - 128)] = (__bf16)v;      // k
        } else {
          const int bb = row >> 11;  // S == 2048
          const int ss = row & 2047;
          out2[((size_t)bb * 128 + (col - 256)) * Sseq + ss] = (__bf16)v;  // vT
        }
      } else if (MODE == 1 || MODE == 3) {
        outF[idx] = resid[idx] + v;
      } else {  // MODE 2: exact GELU
        float gl = 0.5f * v * (1.0f + erff(v * 0.70710678118654752f));
        out0[idx] = (__bf16)gl;
      }
    }
  }
}

// ---------------------------------------------------------------------------
// Flash attention: 1 wave = one (batch, 16-row q tile). 8 waves / block.
// Scores via WMMA (q[16x128] x kT chunks), online softmax with half-wave
// shuffles, P staged through a private LDS slice (s_wait_dscnt ordering),
// then P[16x32] x V[32x128] via WMMA against vT.
// ---------------------------------------------------------------------------
__global__ __launch_bounds__(256) void k_attention(
    const __bf16* __restrict__ Q, const __bf16* __restrict__ Kb,
    const __bf16* __restrict__ Vt, __bf16* __restrict__ H, int Sseq) {
  __shared__ __bf16 Pbuf[8][16 * 32];
  const int wave   = threadIdx.x >> 5;
  const int lane   = threadIdx.x & 31;
  const int half   = lane >> 4;
  const int lanelo = lane & 15;
  const int wg   = blockIdx.x * 8 + wave;
  const int nqt  = Sseq >> 4;
  const int bidx = wg / nqt;
  const int q0   = (wg % nqt) * 16;
  const float scale = 0.088388347648318447f;  // 1/sqrt(128)

  const __bf16* qb = Q  + ((size_t)bidx * Sseq + q0) * 128;
  const __bf16* kb = Kb + (size_t)bidx * Sseq * 128;
  const __bf16* vb = Vt + (size_t)bidx * 128 * Sseq;

  v16bf qf[4];
#pragma unroll
  for (int kk = 0; kk < 4; ++kk) qf[kk] = ld_frag(qb + kk * 32, 128);

  v8f oacc[8] = {};
  float m[8], l[8];
#pragma unroll
  for (int j = 0; j < 8; ++j) { m[j] = -3.0e38f; l[j] = 0.f; }

  __bf16* pw = &Pbuf[wave][0];

  for (int s0 = 0; s0 < Sseq; s0 += 32) {
    // ---- scores: 2 tiles of 16x16, K-dim = 128 (4 WMMA chunks each) ----
    v8f st[2] = {};
#pragma unroll
    for (int tn = 0; tn < 2; ++tn) {
#pragma unroll
      for (int kk = 0; kk < 4; ++kk) {
        v16bf kf = ld_frag(kb + (size_t)(s0 + 16 * tn) * 128 + kk * 32, 128);
        st[tn] = wmma_bf16(qf[kk], kf, st[tn]);
      }
    }
    // ---- online softmax (rows live across 16-lane halves) ----
    float mnew[8], fac[8], rsum[8];
#pragma unroll
    for (int j = 0; j < 8; ++j) {
      float mj = fmaxf(st[0][j], st[1][j]);
#pragma unroll
      for (int off = 1; off < 16; off <<= 1)
        mj = fmaxf(mj, __shfl_xor(mj, off, 32));
      mj *= scale;
      mnew[j] = fmaxf(m[j], mj);
      fac[j]  = __expf(m[j] - mnew[j]);
      rsum[j] = 0.f;
    }
#pragma unroll
    for (int tn = 0; tn < 2; ++tn) {
#pragma unroll
      for (int j = 0; j < 8; ++j) {
        float e = __expf(st[tn][j] * scale - mnew[j]);
        rsum[j] += e;
        pw[(j + 8 * half) * 32 + 16 * tn + lanelo] = (__bf16)e;
      }
    }
#pragma unroll
    for (int j = 0; j < 8; ++j) {
      float r = rsum[j];
#pragma unroll
      for (int off = 1; off < 16; off <<= 1) r += __shfl_xor(r, off, 32);
      l[j] = l[j] * fac[j] + r;
      m[j] = mnew[j];
    }
#pragma unroll
    for (int t = 0; t < 8; ++t)
#pragma unroll
      for (int j = 0; j < 8; ++j) oacc[t][j] *= fac[j];

    // ensure P stores landed before A-fragment reload (CDNA5 split counter)
    asm volatile("s_wait_dscnt 0" ::: "memory");
    v16bf pa = ld_frag(pw, 32);

    // ---- O += P[16x32] x V[32x128] (8 col tiles from vT) ----
#pragma unroll
    for (int t = 0; t < 8; ++t) {
      v16bf vf = ld_frag(vb + (size_t)(16 * t) * Sseq + s0, Sseq);
      oacc[t] = wmma_bf16(pa, vf, oacc[t]);
    }
  }

#pragma unroll
  for (int t = 0; t < 8; ++t) {
#pragma unroll
    for (int j = 0; j < 8; ++j) {
      size_t idx = ((size_t)bidx * Sseq + q0 + j + 8 * half) * 128 +
                   16 * t + lanelo;
      H[idx] = (__bf16)(oacc[t][j] / l[j]);
    }
  }
}

// ---------------------------------------------------------------------------
// Host-side launch
// ---------------------------------------------------------------------------
extern "C" void kernel_launch(void* const* d_in, const int* in_sizes, int n_in,
                              void* d_out, int out_size, void* d_ws,
                              size_t ws_size, hipStream_t stream) {
  (void)in_sizes; (void)n_in; (void)out_size; (void)ws_size;
  const int B = 8, S = 2048, D = 1024, DK = 128, H4 = 4096;
  const int M = B * S;  // 16384

  const float* x    = (const float*)d_in[0];
  const float* ln1g = (const float*)d_in[1];
  const float* ln1b = (const float*)d_in[2];
  const float* Wq   = (const float*)d_in[3];
  const float* bq   = (const float*)d_in[4];
  const float* Wk   = (const float*)d_in[5];
  const float* bk   = (const float*)d_in[6];
  const float* Wv   = (const float*)d_in[7];
  const float* bv   = (const float*)d_in[8];
  const float* Wo   = (const float*)d_in[9];
  const float* bo   = (const float*)d_in[10];
  const float* ln2g = (const float*)d_in[11];
  const float* ln2b = (const float*)d_in[12];
  const float* W1   = (const float*)d_in[13];
  const float* b1   = (const float*)d_in[14];
  const float* W2   = (const float*)d_in[15];
  const float* b2   = (const float*)d_in[16];

  char* ws = (char*)d_ws;
  size_t off = 0;
  auto alloc = [&](size_t bytes) {
    void* p = ws + off;
    off += (bytes + 255) & ~(size_t)255;
    return p;
  };
  float*  xn_f  = (float*) alloc((size_t)M * D * 4);   // xn, then y in-place
  __bf16* xn_b  = (__bf16*)alloc((size_t)M * D * 2);
  __bf16* h_b   = (__bf16*)alloc((size_t)M * D * 2);   // ln2 out
  __bf16* q_b   = (__bf16*)alloc((size_t)M * DK * 2);
  __bf16* k_b   = (__bf16*)alloc((size_t)M * DK * 2);
  __bf16* vT_b  = (__bf16*)alloc((size_t)M * DK * 2);  // [B][128][S]
  __bf16* H_b   = (__bf16*)alloc((size_t)M * DK * 2);
  __bf16* h1_b  = (__bf16*)alloc((size_t)M * H4 * 2);  // gelu out
  __bf16* WqkvT = (__bf16*)alloc((size_t)384 * D * 2);
  __bf16* WoT   = (__bf16*)alloc((size_t)D * DK * 2);
  __bf16* W1T   = (__bf16*)alloc((size_t)H4 * D * 2);
  __bf16* W2T   = (__bf16*)alloc((size_t)D * H4 * 2);
  float*  qkvb  = (float*) alloc(384 * 4);

  // ---- weight prep (fp32 -> bf16, transposed) ----
  int nQ = D * DK;
  k_convT<<<(nQ + 255) / 256, 256, 0, stream>>>(Wq, WqkvT,            D, DK);
  k_convT<<<(nQ + 255) / 256, 256, 0, stream>>>(Wk, WqkvT + 128 * D,  D, DK);
  k_convT<<<(nQ + 255) / 256, 256, 0, stream>>>(Wv, WqkvT + 256 * D,  D, DK);
  k_convT<<<(nQ + 255) / 256, 256, 0, stream>>>(Wo, WoT, DK, D);
  int nM = D * H4;
  k_convT<<<(nM + 255) / 256, 256, 0, stream>>>(W1, W1T, D, H4);
  k_convT<<<(nM + 255) / 256, 256, 0, stream>>>(W2, W2T, H4, D);
  k_cat_bias<<<1, 384, 0, stream>>>(bq, bk, bv, qkvb);

  // ---- LN1: x -> xn (fp32 + bf16) ----
  k_layernorm<<<M, 256, 0, stream>>>(x, ln1g, ln1b, xn_f, xn_b, D);

  // ---- fused QKV projection ----
  k_gemm<0><<<dim3(M / 128, 384 / 128), 256, 0, stream>>>(
      xn_b, WqkvT, M, 384, D, qkvb, nullptr, nullptr, q_b, k_b, vT_b, S);

  // ---- flash attention -> H ----
  k_attention<<<(M / 16) / 8, 256, 0, stream>>>(q_b, k_b, vT_b, H_b, S);

  // ---- output proj + residual: y = xn + H@Wo + bo (in-place into xn_f) ----
  k_gemm<1><<<dim3(M / 128, D / 128), 256, 0, stream>>>(
      H_b, WoT, M, D, DK, bo, xn_f, xn_f, nullptr, nullptr, nullptr, S);

  // ---- LN2: y -> h (bf16) ----
  k_layernorm<<<M, 256, 0, stream>>>(xn_f, ln2g, ln2b, nullptr, h_b, D);

  // ---- MLP1: gelu(h@W1 + b1) -> h1 ----
  k_gemm<2><<<dim3(M / 128, H4 / 128), 256, 0, stream>>>(
      h_b, W1T, M, H4, D, b1, nullptr, nullptr, h1_b, nullptr, nullptr, S);

  // ---- MLP2: out = y + h1@W2 + b2 ----
  k_gemm<3><<<dim3(M / 128, D / 128), 256, 0, stream>>>(
      h1_b, W2T, M, D, H4, b2, xn_f, (float*)d_out, nullptr, nullptr, nullptr,
      S);
}
